// EnhancedTransformer_15350213116361
// MI455X (gfx1250) — matrix operand order repaired and checked by reading, not compiled, not run on hardware
//
#include <hip/hip_runtime.h>
#include <hip/hip_bf16.h>
#include <math.h>

// ---------------- CDNA5 WMMA helpers (wave32, 16x16x32 f16 -> f32) ----------
typedef __attribute__((ext_vector_type(16))) _Float16 v16h;
typedef __attribute__((ext_vector_type(8)))  _Float16 v8h;
typedef __attribute__((ext_vector_type(8)))  float    v8f;
typedef __attribute__((ext_vector_type(4)))  float    v4f;

__device__ __forceinline__ v8f zero8() {
  v8f z = {0.f,0.f,0.f,0.f,0.f,0.f,0.f,0.f};
  return z;
}

__device__ __forceinline__ v8f wmma_f16(v16h a, v16h b, v8f c) {
  // D = A(16x32) * B(32x16) + C ; fp32 accumulate
  return __builtin_amdgcn_wmma_f32_16x16x32_f16(
      /*neg_a=*/false, a, /*neg_b=*/false, b,
      /*c_mod=*/(short)0, c, /*reuse_a=*/false, /*reuse_b=*/false);
}

// ---- CDNA5 async global->LDS copy (ASYNCcnt-tracked, no VGPR round trip) ---
// LDS-aperture flat addresses carry the workgroup-relative LDS byte offset in
// addr[31:0] (ISA 10.2), so truncating a generic shared pointer gives VDST.
__device__ __forceinline__ void async_copy_b128(const void* g, void* lds_dst) {
  unsigned loff = (unsigned)(uintptr_t)lds_dst;
  asm volatile("global_load_async_to_lds_b128 %0, %1, off"
               :: "v"(loff), "v"(g) : "memory");
}
__device__ __forceinline__ void wait_async0() {
  asm volatile("s_wait_asynccnt 0x0" ::: "memory");
}
// copy nelem f16 (nelem % 2048 == 0) from global to LDS, 16B per lane
__device__ __forceinline__ void stage_async_f16(const _Float16* __restrict__ g,
                                                _Float16* l, int nelem, int tid) {
  for (int i = tid * 8; i < nelem; i += 256 * 8)
    async_copy_b128(g + i, l + i);
}
// vectorized f32 -> f16 staging: 2x global b128 loads + 1x ds b128 store
__device__ __forceinline__ void stage_cvt_f16(const float* __restrict__ g,
                                              _Float16* l, int nelem, int tid) {
  for (int i = tid * 8; i < nelem; i += 256 * 8) {
    v4f a = *(const v4f*)(g + i);
    v4f b = *(const v4f*)(g + i + 4);
    v8h h;
    h[0] = (_Float16)a[0]; h[1] = (_Float16)a[1];
    h[2] = (_Float16)a[2]; h[3] = (_Float16)a[3];
    h[4] = (_Float16)b[0]; h[5] = (_Float16)b[1];
    h[6] = (_Float16)b[2]; h[7] = (_Float16)b[3];
    *(v8h*)(l + i) = h;
  }
}

// A fragment, A (16x32, MxK) row-major in memory, stride lda.
// Layout: lanes0-15 rows M, K=0..15 (2 per VGPR); lanes16-31 K=16..31.
__device__ __forceinline__ v16h frag_A_rowmajor(const _Float16* A, int lda, int lane) {
  int m = lane & 15, hi = (lane >> 4) & 1;
  v16h f;
  const _Float16* p = A + m * lda;
#pragma unroll
  for (int v = 0; v < 8; ++v) {
    int k0 = hi * 16 + 2 * v;
    f[2 * v]     = p[k0];
    f[2 * v + 1] = p[k0 + 1];
  }
  return f;
}

// A fragment where memory holds A transposed: A[m][k] = mem[k*lda + m]
__device__ __forceinline__ v16h frag_A_kmajor(const _Float16* A, int lda, int lane) {
  int m = lane & 15, hi = (lane >> 4) & 1;
  v16h f;
#pragma unroll
  for (int v = 0; v < 8; ++v) {
    int k0 = hi * 16 + 2 * v;
    f[2 * v]     = A[k0 * lda + m];
    f[2 * v + 1] = A[(k0 + 1) * lda + m];
  }
  return f;
}

// B fragment, B (32x16, KxN) with memory k-major: B[k][n] = mem[k*ldb + n].
// Layout (contiguous k, same as the A frags): lanes0-15 hold K=0..15 (2 per VGPR), lanes16-31 hold K=16..31.
__device__ __forceinline__ v16h frag_B_kmajor(const _Float16* Bp, int ldb, int lane) {
  int n = lane & 15, hi = (lane >> 4) & 1;
  v16h f;
#pragma unroll
  for (int v = 0; v < 8; ++v) {
    int k0 = hi * 16 + 2 * v;
    f[2 * v]     = Bp[k0 * ldb + n];
    f[2 * v + 1] = Bp[(k0 + 1) * ldb + n];
  }
  return f;
}

// B fragment for X @ W^T: B[k][n] = W[n*ldw + k], W is (N x K) row-major.
__device__ __forceinline__ v16h frag_B_nmajor(const _Float16* Wp, int ldw, int lane) {
  int n = lane & 15, hi = (lane >> 4) & 1;
  v16h f;
  const _Float16* p = Wp + n * ldw;
#pragma unroll
  for (int v = 0; v < 8; ++v) {
    int k0 = hi * 16 + 2 * v;
    f[2 * v]     = p[k0];
    f[2 * v + 1] = p[k0 + 1];
  }
  return f;
}

// ---------------------------------------------------------------------------
// Problem constants
#define BN 8
#define SN 2048
#define DN 128
#define HN 8
#define WN 64
#define FFN 512

// ---------------- weight conversion ----------------------------------------
__global__ void k_f32_to_f16(const float* __restrict__ src, _Float16* __restrict__ dst, int n) {
  int i = blockIdx.x * blockDim.x + threadIdx.x;
  if (i < n) dst[i] = (_Float16)src[i];
}

// ---------------- kernel 1: windowed causal MHA + residual + LN1 -----------
// grid = 256 (= B * S/W) blocks, 256 threads (8 waves); one 64-token window.
__global__ __launch_bounds__(256) void k_win_attn(
    const float* __restrict__ x,
    const _Float16* __restrict__ Win, const float* __restrict__ bin,
    const _Float16* __restrict__ Wout, const float* __restrict__ bout,
    const float* __restrict__ g1, const float* __restrict__ b1,
    float* __restrict__ y1) {
  __shared__ __align__(16) char smem[81920];
  _Float16* Xh  = (_Float16*)(smem);            // 64x128 f16 (phase 1)
  _Float16* Qh  = (_Float16*)(smem + 16384);    // 64x128
  _Float16* Kh  = (_Float16*)(smem + 32768);    // 64x128
  _Float16* Vh  = (_Float16*)(smem + 49152);    // 64x128
  _Float16* Pb  = (_Float16*)(smem);            // aliases Xh: 8 waves x 16x64
  _Float16* AOh = (_Float16*)(smem + 65536);    // 64x128
  float*    OutF= (float*)(smem + 16384);       // aliases Qh/Kh: 64x128 f32

  const int tid = threadIdx.x;
  const int wave = tid >> 5, lane = tid & 31;
  const int n = lane & 15, hi = (lane >> 4) & 1;
  const long row0 = (long)blockIdx.x * WN;      // flattened token row base

  stage_cvt_f16(x + row0 * DN, Xh, WN * DN, tid);
  __syncthreads();

  // --- QKV projection: 3 x (64x128) outputs = 96 tiles, 12 per wave -------
  for (int ii = 0; ii < 12; ++ii) {
    int t = wave * 12 + ii;
    int which = t >> 5;           // 0=Q, 1=K, 2=V
    int rem = t & 31;
    int rt = rem >> 3, ct = rem & 7;
    v8f acc = zero8();
#pragma unroll
    for (int kk = 0; kk < 4; ++kk) {
      v16h a = frag_A_rowmajor(Xh + rt * 16 * DN + kk * 32, DN, lane);
      v16h b = frag_B_nmajor(Win + (which * DN + ct * 16) * DN + kk * 32, DN, lane);
      acc = wmma_f16(a, b, acc);
    }
    _Float16* Dst = (which == 0) ? Qh : (which == 1) ? Kh : Vh;
#pragma unroll
    for (int v = 0; v < 8; ++v) {
      int r = rt * 16 + v + 8 * hi, c = ct * 16 + n;
      Dst[r * DN + c] = (_Float16)(acc[v] + bin[which * DN + c]);
    }
  }
  __syncthreads();  // Q/K/V ready; Xh dead -> Pb may alias it now

  // --- attention: wave `wave` owns head h = wave ---------------------------
  {
    const int h = wave;
    _Float16* Pw = Pb + wave * (16 * 64);
    for (int rt = 0; rt < 4; ++rt) {
      // A = Q_h rows (16 x K), K padded 16 -> 32 with zeros
      v16h aq;
      {
        const _Float16* qrow = Qh + (rt * 16 + (lane & 15)) * DN + h * 16;
#pragma unroll
        for (int v = 0; v < 8; ++v) {
          if (hi == 0) {
            int k0 = 2 * v;                // lanes 0-15: K=0..15 valid
            aq[2 * v]     = qrow[k0];
            aq[2 * v + 1] = qrow[k0 + 1];
          } else {
            aq[2 * v]     = (_Float16)0.f; // lanes 16-31: K=16..31 padded
            aq[2 * v + 1] = (_Float16)0.f;
          }
        }
      }
      // scores tiles c = 0..3 : B[k][j] = K[(16c+j)][h*16+k] (K<16 valid)
      v8f sc[4];
      for (int c = 0; c < 4; ++c) {
        v16h bk;
#pragma unroll
        for (int v = 0; v < 8; ++v) {
          if (hi == 0) {
            int k0 = 2 * v;                // lanes 0-15: K=0..15 valid
            bk[2 * v]     = Kh[(c * 16 + n) * DN + h * 16 + k0];
            bk[2 * v + 1] = Kh[(c * 16 + n) * DN + h * 16 + k0 + 1];
          } else {                          // lanes 16-31: K=16..31 padded
            bk[2 * v]     = (_Float16)0.f;
            bk[2 * v + 1] = (_Float16)0.f;
          }
        }
        v8f z = zero8();
        sc[c] = wmma_f16(aq, bk, z);
      }
      // causal mask + softmax per row (16-lane half-wave reduction)
#pragma unroll
      for (int v = 0; v < 8; ++v) {
        int i_row = rt * 16 + v + 8 * hi;
        float mx = -3.0e38f;
#pragma unroll
        for (int c = 0; c < 4; ++c) {
          int j = c * 16 + n;
          float s = sc[c][v] * 0.25f;      // 1/sqrt(16)
          if (j > i_row) s = -3.0e38f;
          sc[c][v] = s;
          mx = fmaxf(mx, s);
        }
        for (int off = 1; off < 16; off <<= 1)
          mx = fmaxf(mx, __shfl_xor(mx, off, 32));
        float sum = 0.f;
#pragma unroll
        for (int c = 0; c < 4; ++c) {
          float e = __expf(sc[c][v] - mx);
          sc[c][v] = e;
          sum += e;
        }
        for (int off = 1; off < 16; off <<= 1)
          sum += __shfl_xor(sum, off, 32);
        float inv = 1.f / sum;
#pragma unroll
        for (int c = 0; c < 4; ++c)
          Pw[(v + 8 * hi) * 64 + c * 16 + n] = (_Float16)(sc[c][v] * inv);
      }
      // out tile = P(16x64) * V_h(64x16)
      v8f o = zero8();
#pragma unroll
      for (int kk = 0; kk < 2; ++kk) {
        v16h ap = frag_A_rowmajor(Pw + kk * 32, 64, lane);
        v16h bv = frag_B_kmajor(Vh + kk * 32 * DN + h * 16, DN, lane);
        o = wmma_f16(ap, bv, o);
      }
#pragma unroll
      for (int v = 0; v < 8; ++v)
        AOh[(rt * 16 + v + 8 * hi) * DN + h * 16 + n] = (_Float16)o[v];
    }
  }
  __syncthreads();

  // --- output projection + bias + residual ---------------------------------
  for (int cc = 0; cc < 4; ++cc) {
    int rt = wave & 3;
    int ct = (wave >> 2) * 4 + cc;
    v8f acc = zero8();
#pragma unroll
    for (int kk = 0; kk < 4; ++kk) {
      v16h a = frag_A_rowmajor(AOh + rt * 16 * DN + kk * 32, DN, lane);
      v16h b = frag_B_nmajor(Wout + (ct * 16) * DN + kk * 32, DN, lane);
      acc = wmma_f16(a, b, acc);
    }
#pragma unroll
    for (int v = 0; v < 8; ++v) {
      int r = rt * 16 + v + 8 * hi, c = ct * 16 + n;
      OutF[r * DN + c] = acc[v] + bout[c] + x[(row0 + r) * DN + c];
    }
  }
  __syncthreads();

  // --- LayerNorm 1 ----------------------------------------------------------
  if (tid < WN) {
    const float* pr = OutF + tid * DN;
    float mu = 0.f;
    for (int j = 0; j < DN; ++j) mu += pr[j];
    mu *= (1.f / DN);
    float var = 0.f;
    for (int j = 0; j < DN; ++j) { float d = pr[j] - mu; var += d * d; }
    var *= (1.f / DN);
    float rs = rsqrtf(var + 1e-5f);
    for (int j = 0; j < DN; ++j)
      y1[(row0 + tid) * DN + j] = (pr[j] - mu) * rs * g1[j] + b1[j];
  }
}

// ---------------- kernel 2: FFN + residual + LN2 ---------------------------
__global__ __launch_bounds__(256) void k_ffn(
    const float* __restrict__ y1,
    const _Float16* __restrict__ W1, const float* __restrict__ b1,
    const _Float16* __restrict__ W2, const float* __restrict__ b2,
    const float* __restrict__ g2, const float* __restrict__ bb2,
    float* __restrict__ out) {
  __shared__ __align__(16) char smem[114688];
  _Float16* Xh  = (_Float16*)(smem);            // 64x128
  _Float16* Hh  = (_Float16*)(smem + 16384);    // 64x512
  float*    OutF= (float*)(smem + 81920);       // 64x128 f32
  const int tid = threadIdx.x, wave = tid >> 5, lane = tid & 31;
  const int n = lane & 15, hi = (lane >> 4) & 1;
  const long row0 = (long)blockIdx.x * WN;

  stage_cvt_f16(y1 + row0 * DN, Xh, WN * DN, tid);
  __syncthreads();

  // GEMM1: 64x512 = 128 tiles, 16 per wave; exact-erf GELU epilogue
  for (int ii = 0; ii < 16; ++ii) {
    int rt = wave & 3;
    int ct = (wave >> 2) * 16 + ii;
    v8f acc = zero8();
#pragma unroll
    for (int kk = 0; kk < 4; ++kk) {
      v16h a = frag_A_rowmajor(Xh + rt * 16 * DN + kk * 32, DN, lane);
      v16h b = frag_B_nmajor(W1 + (ct * 16) * DN + kk * 32, DN, lane);
      acc = wmma_f16(a, b, acc);
    }
#pragma unroll
    for (int v = 0; v < 8; ++v) {
      int r = rt * 16 + v + 8 * hi, c = ct * 16 + n;
      float val = acc[v] + b1[c];
      val = 0.5f * val * (1.0f + erff(val * 0.70710678118654752f));
      Hh[r * FFN + c] = (_Float16)val;
    }
  }
  __syncthreads();

  // GEMM2: 64x128, + bias + residual(y1)
  for (int cc = 0; cc < 4; ++cc) {
    int rt = wave & 3;
    int ct = (wave >> 2) * 4 + cc;
    v8f acc = zero8();
#pragma unroll
    for (int kk = 0; kk < 16; ++kk) {
      v16h a = frag_A_rowmajor(Hh + rt * 16 * FFN + kk * 32, FFN, lane);
      v16h b = frag_B_nmajor(W2 + (ct * 16) * FFN + kk * 32, FFN, lane);
      acc = wmma_f16(a, b, acc);
    }
#pragma unroll
    for (int v = 0; v < 8; ++v) {
      int r = rt * 16 + v + 8 * hi, c = ct * 16 + n;
      OutF[r * DN + c] = acc[v] + b2[c] + y1[(row0 + r) * DN + c];
    }
  }
  __syncthreads();

  if (tid < WN) {
    const float* pr = OutF + tid * DN;
    float mu = 0.f;
    for (int j = 0; j < DN; ++j) mu += pr[j];
    mu *= (1.f / DN);
    float var = 0.f;
    for (int j = 0; j < DN; ++j) { float d = pr[j] - mu; var += d * d; }
    var *= (1.f / DN);
    float rs = rsqrtf(var + 1e-5f);
    for (int j = 0; j < DN; ++j)
      out[(row0 + tid) * DN + j] = (pr[j] - mu) * rs * g2[j] + bb2[j];
  }
}

// ---------------- kernel 3: se/te projection + cosine normalize ------------
// grid = 512: blocks [0,256) -> spatial, [256,512) -> temporal.
__global__ __launch_bounds__(256) void k_proj_norm(
    const float* __restrict__ spatial, const float* __restrict__ temporal,
    const _Float16* __restrict__ Ws, const float* __restrict__ bs,
    const _Float16* __restrict__ Wt, const float* __restrict__ bt,
    _Float16* __restrict__ seh, _Float16* __restrict__ teh,
    _Float16* __restrict__ snh, _Float16* __restrict__ tnh) {
  __shared__ __align__(16) char smem[49152];
  _Float16* Xh  = (_Float16*)(smem);            // 64x128
  float*    OutF= (float*)(smem + 16384);       // 64x128 f32
  const int tid = threadIdx.x, wave = tid >> 5, lane = tid & 31;
  const int n = lane & 15, hi = (lane >> 4) & 1;
  const int sel = blockIdx.x >> 8;
  const float* in = sel ? temporal : spatial;
  const _Float16* Wm = sel ? Wt : Ws;
  const float* bm = sel ? bt : bs;
  _Float16* eo = sel ? teh : seh;
  _Float16* no = sel ? tnh : snh;
  const long row0 = (long)(blockIdx.x & 255) * WN;

  stage_cvt_f16(in + row0 * DN, Xh, WN * DN, tid);
  __syncthreads();

  for (int cc = 0; cc < 4; ++cc) {
    int rt = wave & 3;
    int ct = (wave >> 2) * 4 + cc;
    v8f acc = zero8();
#pragma unroll
    for (int kk = 0; kk < 4; ++kk) {
      v16h a = frag_A_rowmajor(Xh + rt * 16 * DN + kk * 32, DN, lane);
      v16h b = frag_B_nmajor(Wm + (ct * 16) * DN + kk * 32, DN, lane);
      acc = wmma_f16(a, b, acc);
    }
#pragma unroll
    for (int v = 0; v < 8; ++v) {
      int r = rt * 16 + v + 8 * hi, c = ct * 16 + n;
      OutF[r * DN + c] = acc[v] + bm[c];
    }
  }
  __syncthreads();

  if (tid < WN) {
    const float* pr = OutF + tid * DN;
    float ss = 0.f;
    for (int j = 0; j < DN; ++j) ss += pr[j] * pr[j];
    float inv = 1.f / fmaxf(sqrtf(ss), 1e-8f);
    for (int j = 0; j < DN; ++j) {
      eo[(row0 + tid) * DN + j] = (_Float16)pr[j];
      no[(row0 + tid) * DN + j] = (_Float16)(pr[j] * inv);
    }
  }
}

// ---------------- kernel 4: per-batch Gram M_b = sn^T tn (128x128) ---------
// grid = 8 (one batch per block); wave w owns dim-rows [16w, 16w+16).
// Double-buffered async global->LDS staging overlapped with WMMA.
__global__ __launch_bounds__(256) void k_gram(
    const _Float16* __restrict__ snh, const _Float16* __restrict__ tnh,
    _Float16* __restrict__ Mh) {
  __shared__ __align__(16) char smem[32768];
  // buffer j: S at j*16384, T at j*16384 + 8192 (each 32x128 f16)
  const int tid = threadIdx.x, wave = tid >> 5, lane = tid & 31;
  const int n = lane & 15, hi = (lane >> 4) & 1;
  const int b = blockIdx.x;

  v8f acc[8];
#pragma unroll
  for (int q = 0; q < 8; ++q) acc[q] = zero8();

  const int NKC = SN / 32;  // 64 chunks
  // prologue: stage chunk 0 into buffer 0
  {
    long base = (long)b * SN * DN;
    stage_async_f16(snh + base, (_Float16*)(smem), 32 * DN, tid);
    stage_async_f16(tnh + base, (_Float16*)(smem + 8192), 32 * DN, tid);
  }
  for (int kc = 0; kc < NKC; ++kc) {
    wait_async0();       // my async loads for chunk kc have landed in LDS
    __syncthreads();     // everyone's have
    int cur = kc & 1, nxt = cur ^ 1;
    _Float16* Sh = (_Float16*)(smem + cur * 16384);
    _Float16* Th = (_Float16*)(smem + cur * 16384 + 8192);
    if (kc + 1 < NKC) {  // prefetch next chunk while computing this one
      long base = ((long)b * SN + (kc + 1) * 32) * DN;
      stage_async_f16(snh + base, (_Float16*)(smem + nxt * 16384), 32 * DN, tid);
      stage_async_f16(tnh + base, (_Float16*)(smem + nxt * 16384 + 8192), 32 * DN, tid);
    }
    // A = sn^T tile (dims x tokens): A[p][j] = Sh[j*128 + 16*wave + p]
    v16h a = frag_A_kmajor(Sh + wave * 16, DN, lane);
#pragma unroll
    for (int q = 0; q < 8; ++q) {
      v16h bf = frag_B_kmajor(Th + q * 16, DN, lane);
      acc[q] = wmma_f16(a, bf, acc[q]);
    }
    __syncthreads();     // all waves done reading `cur` before it is refilled
  }
#pragma unroll
  for (int q = 0; q < 8; ++q)
#pragma unroll
    for (int v = 0; v < 8; ++v)
      Mh[((long)b * DN + wave * 16 + v + 8 * hi) * DN + q * 16 + n] =
          (_Float16)acc[q][v];
}

// ---------------- kernel 5: sim[b,i] = sn_i . (M_b tn_i) / S ---------------
__global__ __launch_bounds__(256) void k_sim(
    const _Float16* __restrict__ snh, const _Float16* __restrict__ tnh,
    const _Float16* __restrict__ Mh, float* __restrict__ sim) {
  __shared__ __align__(16) char smem[49152];
  _Float16* Th = (_Float16*)(smem);            // 64x128 (tn rows)
  float*    UF = (float*)(smem + 16384);       // 64x128 f32 (U = Tn*M^T)
  const int tid = threadIdx.x, wave = tid >> 5, lane = tid & 31;
  const int n = lane & 15, hi = (lane >> 4) & 1;
  const long row0 = (long)blockIdx.x * WN;
  const int b = (int)(row0 / SN);
  const _Float16* Mb = Mh + (long)b * DN * DN;

  stage_async_f16(tnh + row0 * DN, Th, WN * DN, tid);
  wait_async0();
  __syncthreads();

  // U[i][p] = sum_q Tn[i][q] * M[p][q]  -> B[k=q][n=p] = M[p*128+q] (n-major)
  for (int cc = 0; cc < 4; ++cc) {
    int rt = wave & 3;
    int ct = (wave >> 2) * 4 + cc;
    v8f acc = zero8();
#pragma unroll
    for (int kk = 0; kk < 4; ++kk) {
      v16h a = frag_A_rowmajor(Th + rt * 16 * DN + kk * 32, DN, lane);
      v16h bf = frag_B_nmajor(Mb + (ct * 16) * DN + kk * 32, DN, lane);
      acc = wmma_f16(a, bf, acc);
    }
#pragma unroll
    for (int v = 0; v < 8; ++v)
      UF[(rt * 16 + v + 8 * hi) * DN + ct * 16 + n] = acc[v];
  }
  __syncthreads();

  if (tid < WN) {
    const float* ur = UF + tid * DN;
    const _Float16* sr = snh + (row0 + tid) * DN;
    float dot = 0.f;
    for (int j = 0; j < DN; ++j) dot += ur[j] * (float)sr[j];
    sim[row0 + tid] = dot * (1.0f / (float)SN);
  }
}

// ---------------- kernel 6: interaction MHA QKV projection -----------------
__global__ __launch_bounds__(256) void k_int_qkv(
    const _Float16* __restrict__ seh, const _Float16* __restrict__ teh,
    const _Float16* __restrict__ Wi, const float* __restrict__ bi,
    _Float16* __restrict__ qih, _Float16* __restrict__ kih,
    _Float16* __restrict__ vih) {
  __shared__ __align__(16) char smem[32768];
  _Float16* Sh = (_Float16*)(smem);            // 64x128 (se rows)
  _Float16* Th = (_Float16*)(smem + 16384);    // 64x128 (te rows)
  const int tid = threadIdx.x, wave = tid >> 5, lane = tid & 31;
  const int n = lane & 15, hi = (lane >> 4) & 1;
  const long row0 = (long)blockIdx.x * WN;

  stage_async_f16(seh + row0 * DN, Sh, WN * DN, tid);
  stage_async_f16(teh + row0 * DN, Th, WN * DN, tid);
  wait_async0();
  __syncthreads();

  for (int ii = 0; ii < 12; ++ii) {
    int t = wave * 12 + ii;
    int which = t >> 5;          // 0=Q(from se), 1=K(te), 2=V(te)
    int rem = t & 31;
    int rt = rem >> 3, ct = rem & 7;
    const _Float16* A = (which == 0) ? Sh : Th;
    v8f acc = zero8();
#pragma unroll
    for (int kk = 0; kk < 4; ++kk) {
      v16h a = frag_A_rowmajor(A + rt * 16 * DN + kk * 32, DN, lane);
      v16h bf = frag_B_nmajor(Wi + (which * DN + ct * 16) * DN + kk * 32, DN, lane);
      acc = wmma_f16(a, bf, acc);
    }
    _Float16* Dst = (which == 0) ? qih : (which == 1) ? kih : vih;
#pragma unroll
    for (int v = 0; v < 8; ++v) {
      int r = rt * 16 + v + 8 * hi, c = ct * 16 + n;
      Dst[(row0 + r) * DN + c] = (_Float16)(acc[v] + bi[which * DN + c]);
    }
  }
}

// ---------------- kernel 7: tiny 8x8 interaction attention (len = B) -------
// one thread per (s, head); loops queries i = 0..7.
__global__ __launch_bounds__(256) void k_int_attn(
    const _Float16* __restrict__ qih, const _Float16* __restrict__ kih,
    const _Float16* __restrict__ vih, _Float16* __restrict__ aoh) {
  int idx = blockIdx.x * blockDim.x + threadIdx.x;
  if (idx >= SN * HN) return;
  int s = idx >> 3, h = idx & 7;
  int d0 = h * 16;
  for (int i = 0; i < BN; ++i) {
    float q[16];
    const _Float16* qp = qih + ((long)i * SN + s) * DN + d0;
    for (int d = 0; d < 16; ++d) q[d] = (float)qp[d];
    float sc[BN];
    float mx = -3.0e38f;
    for (int j = 0; j < BN; ++j) {
      const _Float16* kp = kih + ((long)j * SN + s) * DN + d0;
      float dot = 0.f;
      for (int d = 0; d < 16; ++d) dot += q[d] * (float)kp[d];
      sc[j] = dot * 0.25f;
      mx = fmaxf(mx, sc[j]);
    }
    float sum = 0.f;
    for (int j = 0; j < BN; ++j) { sc[j] = __expf(sc[j] - mx); sum += sc[j]; }
    float inv = 1.f / sum;
    float o[16];
    for (int d = 0; d < 16; ++d) o[d] = 0.f;
    for (int j = 0; j < BN; ++j) {
      const _Float16* vp = vih + ((long)j * SN + s) * DN + d0;
      float p = sc[j] * inv;
      for (int d = 0; d < 16; ++d) o[d] += p * (float)vp[d];
    }
    _Float16* op = aoh + ((long)i * SN + s) * DN + d0;
    for (int d = 0; d < 16; ++d) op[d] = (_Float16)o[d];
  }
}

// ---------------- kernel 8: inter out-proj + final fused epilogue ----------
// out[b,s,:] += sim[b,s] * (ao @ Wo^T + bo)
__global__ __launch_bounds__(256) void k_int_out(
    const _Float16* __restrict__ aoh, const _Float16* __restrict__ Wo,
    const float* __restrict__ bo, const float* __restrict__ sim,
    float* __restrict__ out) {
  __shared__ __align__(16) char smem[16384];
  _Float16* Ah = (_Float16*)(smem);            // 64x128
  const int tid = threadIdx.x, wave = tid >> 5, lane = tid & 31;
  const int n = lane & 15, hi = (lane >> 4) & 1;
  const long row0 = (long)blockIdx.x * WN;

  stage_async_f16(aoh + row0 * DN, Ah, WN * DN, tid);
  wait_async0();
  __syncthreads();

  for (int cc = 0; cc < 4; ++cc) {
    int rt = wave & 3;
    int ct = (wave >> 2) * 4 + cc;
    v8f acc = zero8();
#pragma unroll
    for (int kk = 0; kk < 4; ++kk) {
      v16h a = frag_A_rowmajor(Ah + rt * 16 * DN + kk * 32, DN, lane);
      v16h bf = frag_B_nmajor(Wo + (ct * 16) * DN + kk * 32, DN, lane);
      acc = wmma_f16(a, bf, acc);
    }
#pragma unroll
    for (int v = 0; v < 8; ++v) {
      int r = rt * 16 + v + 8 * hi, c = ct * 16 + n;
      long o = (row0 + r) * DN + c;
      out[o] = out[o] + sim[row0 + r] * (acc[v] + bo[c]);
    }
  }
}

// ---------------------------------------------------------------------------
extern "C" void kernel_launch(void* const* d_in, const int* in_sizes, int n_in,
                              void* d_out, int out_size, void* d_ws, size_t ws_size,
                              hipStream_t stream) {
  (void)in_sizes; (void)n_in; (void)out_size; (void)ws_size;
  const float* x        = (const float*)d_in[0];
  const float* spatial  = (const float*)d_in[1];
  const float* temporal = (const float*)d_in[2];
  const float* lw_in_w  = (const float*)d_in[3];
  const float* lw_in_b  = (const float*)d_in[4];
  const float* lw_out_w = (const float*)d_in[5];
  const float* lw_out_b = (const float*)d_in[6];
  const float* spat_w   = (const float*)d_in[7];
  const float* spat_b   = (const float*)d_in[8];
  const float* temp_w   = (const float*)d_in[9];
  const float* temp_b   = (const float*)d_in[10];
  const float* int_in_w = (const float*)d_in[11];
  const float* int_in_b = (const float*)d_in[12];
  const float* int_out_w= (const float*)d_in[13];
  const float* int_out_b= (const float*)d_in[14];
  const float* ffn_w1   = (const float*)d_in[15];
  const float* ffn_b1   = (const float*)d_in[16];
  const float* ffn_w2   = (const float*)d_in[17];
  const float* ffn_b2   = (const float*)d_in[18];
  const float* ln1_g    = (const float*)d_in[19];
  const float* ln1_b    = (const float*)d_in[20];
  const float* ln2_g    = (const float*)d_in[21];
  const float* ln2_b    = (const float*)d_in[22];

  // ---- workspace layout (bytes) ----
  char* ws = (char*)d_ws;
  size_t off = 0;
  _Float16* Wh_lw_in  = (_Float16*)(ws + off); off += 3 * DN * DN * 2;   // 384x128
  _Float16* Wh_lw_out = (_Float16*)(ws + off); off += DN * DN * 2;
  _Float16* Wh_spat   = (_Float16*)(ws + off); off += DN * DN * 2;
  _Float16* Wh_temp   = (_Float16*)(ws + off); off += DN * DN * 2;
  _Float16* Wh_int_in = (_Float16*)(ws + off); off += 3 * DN * DN * 2;
  _Float16* Wh_int_out= (_Float16*)(ws + off); off += DN * DN * 2;
  _Float16* Wh_ffn1   = (_Float16*)(ws + off); off += FFN * DN * 2;      // 512x128
  _Float16* Wh_ffn2   = (_Float16*)(ws + off); off += DN * FFN * 2;      // 128x512
  off = (off + 255) & ~(size_t)255;
  float*    y1   = (float*)(ws + off);    off += (size_t)BN * SN * DN * 4;
  _Float16* seh  = (_Float16*)(ws + off); off += (size_t)BN * SN * DN * 2;
  _Float16* teh  = (_Float16*)(ws + off); off += (size_t)BN * SN * DN * 2;
  _Float16* snh  = (_Float16*)(ws + off); off += (size_t)BN * SN * DN * 2;
  _Float16* tnh  = (_Float16*)(ws + off); off += (size_t)BN * SN * DN * 2;
  _Float16* Mh   = (_Float16*)(ws + off); off += (size_t)BN * DN * DN * 2;
  float*    simf = (float*)(ws + off);    off += (size_t)BN * SN * 4;
  _Float16* qih  = (_Float16*)(ws + off); off += (size_t)BN * SN * DN * 2;
  _Float16* kih  = (_Float16*)(ws + off); off += (size_t)BN * SN * DN * 2;
  _Float16* vih  = (_Float16*)(ws + off); off += (size_t)BN * SN * DN * 2;
  _Float16* aoh  = (_Float16*)(ws + off); off += (size_t)BN * SN * DN * 2;

  // ---- weight conversion (f32 -> f16) ----
  {
    struct { const float* s; _Float16* d; int n; } cv[8] = {
      { lw_in_w,  Wh_lw_in,  3 * DN * DN },
      { lw_out_w, Wh_lw_out, DN * DN },
      { spat_w,   Wh_spat,   DN * DN },
      { temp_w,   Wh_temp,   DN * DN },
      { int_in_w, Wh_int_in, 3 * DN * DN },
      { int_out_w,Wh_int_out,DN * DN },
      { ffn_w1,   Wh_ffn1,   FFN * DN },
      { ffn_w2,   Wh_ffn2,   DN * FFN },
    };
    for (int i = 0; i < 8; ++i)
      k_f32_to_f16<<<(cv[i].n + 255) / 256, 256, 0, stream>>>(cv[i].s, cv[i].d, cv[i].n);
  }

  const int nchunks = BN * SN / WN;  // 256

  // transformer trunk
  k_win_attn<<<nchunks, 256, 0, stream>>>(x, Wh_lw_in, lw_in_b, Wh_lw_out,
                                          lw_out_b, ln1_g, ln1_b, y1);
  k_ffn<<<nchunks, 256, 0, stream>>>(y1, Wh_ffn1, ffn_b1, Wh_ffn2, ffn_b2,
                                     ln2_g, ln2_b, (float*)d_out);

  // spatio-temporal interaction
  k_proj_norm<<<2 * nchunks, 256, 0, stream>>>(spatial, temporal, Wh_spat, spat_b,
                                               Wh_temp, temp_b, seh, teh, snh, tnh);
  k_gram<<<BN, 256, 0, stream>>>(snh, tnh, Mh);
  k_sim<<<nchunks, 256, 0, stream>>>(snh, tnh, Mh, simf);
  k_int_qkv<<<nchunks, 256, 0, stream>>>(seh, teh, Wh_int_in, int_in_b, qih, kih, vih);
  k_int_attn<<<(SN * HN + 255) / 256, 256, 0, stream>>>(qih, kih, vih, aoh);
  k_int_out<<<nchunks, 256, 0, stream>>>(aoh, Wh_int_out, int_out_b, simf,
                                         (float*)d_out);
}
